// Head_21947282883000
// MI455X (gfx1250) — compile-verified, matrix operand order checked by reading, not run on hardware
//
#include <hip/hip_runtime.h>
#include <hip/hip_bf16.h>

typedef __attribute__((ext_vector_type(16))) _Float16 v16h;
typedef __attribute__((ext_vector_type(8)))  float    v8f;

constexpr int BN = 4, TN = 4096, CN = 768, HN = 64;
constexpr float SM_SCALE = 0.03608439182435161f;  // 1/sqrt(768) -- reference scales by C**-0.5
constexpr float NEG_BIG  = -3.0e38f;

// ---------------------------------------------------------------------------
// D = A*B + C, f16 inputs, f32 accumulate (v_wmma_f32_16x16x32_f16)
__device__ __forceinline__ v8f wmma16(v16h a, v16h b, v8f c) {
  return __builtin_amdgcn_wmma_f32_16x16x32_f16(false, a, false, b, (short)0, c,
                                                false, false);
}

// Load a 16x32 f16 tile (row-major, row stride ld) into the WMMA A/B register
// layout for wave32:
//   lanes 0-15 : row = lane,    halves = cols {0..7} then {16..23}
//   lanes 16-31: row = lane-16, halves = cols {8..15} then {24..31}
__device__ __forceinline__ v16h load_tile_h(const _Float16* base, int ld, int lane) {
  const int r  = lane & 15;
  const int c0 = (lane & 16) ? 8 : 0;
  const _Float16* p = base + r * ld + c0;
  v16h o;
#pragma unroll
  for (int i = 0; i < 8; ++i) { o[i] = p[i]; o[8 + i] = p[16 + i]; }
  return o;
}

// Same tile layout but source is f32 (converted to f16 on the fly).
__device__ __forceinline__ v16h load_tile_f(const float* base, int ld, int lane) {
  const int r  = lane & 15;
  const int c0 = (lane & 16) ? 8 : 0;
  const float* p = base + r * ld + c0;
  v16h o;
#pragma unroll
  for (int i = 0; i < 8; ++i) {
    o[i]     = (_Float16)p[i];
    o[8 + i] = (_Float16)p[16 + i];
  }
  return o;
}

// ---------------------------------------------------------------------------
// Kernel 1: transpose + convert weights to f16.  wt layout: [3][H][C] with
// w=0 -> Wq^T, w=1 -> Wk^T, w=2 -> Wv^T  (row h, col c: contiguous in c).
__global__ void __launch_bounds__(256)
cvt_weights(const float* __restrict__ Wq, const float* __restrict__ Wk,
            const float* __restrict__ Wv, _Float16* __restrict__ wt) {
  int idx = blockIdx.x * blockDim.x + threadIdx.x;
  if (idx >= 3 * CN * HN) return;
  int w   = idx / (CN * HN);
  int rem = idx - w * (CN * HN);
  int h   = rem / CN;
  int c   = rem - h * CN;
  const float* W = (w == 0) ? Wq : (w == 1) ? Wk : Wv;
  wt[idx] = (_Float16)W[c * HN + h];
}

// ---------------------------------------------------------------------------
// Kernel 2: QKV projection.  [B*T, C] x [C, H] (x3) with WMMA, f32 accum.
// One wave owns 16 rows; 8 waves per block -> 128 rows per block.
// Outputs: q16/k16 row-major [B*T, H] f16; v transposed vt16 [B][H][T] f16.
__global__ void __launch_bounds__(256)
qkv_proj(const float* __restrict__ x, const _Float16* __restrict__ wt,
         _Float16* __restrict__ q16, _Float16* __restrict__ k16,
         _Float16* __restrict__ vt16) {
  const int lane    = threadIdx.x & 31;
  const int wave    = threadIdx.x >> 5;
  const int rowbase = blockIdx.x * 128 + wave * 16;

  v8f acc[12];  // [w][ntile], w in {q,k,v}, ntile covers H in 4x16
#pragma unroll
  for (int i = 0; i < 12; ++i) acc[i] = (v8f)0.0f;

  for (int kc = 0; kc < CN; kc += 32) {
    v16h a = load_tile_f(x + (size_t)rowbase * CN + kc, CN, lane);
#pragma unroll
    for (int w = 0; w < 3; ++w) {
#pragma unroll
      for (int j = 0; j < 4; ++j) {
        v16h b = load_tile_h(wt + (size_t)w * HN * CN + (j * 16) * CN + kc, CN, lane);
        acc[w * 4 + j] = wmma16(a, b, acc[w * 4 + j]);
      }
    }
  }

  // C/D layout: element r of acc -> row (r + 8*(lane/16)), col (lane&15).
  const int n    = lane & 15;
  const int r8   = (lane >> 4) * 8;
#pragma unroll
  for (int j = 0; j < 4; ++j) {
#pragma unroll
    for (int r = 0; r < 8; ++r) {
      const int row = rowbase + r + r8;
      const int h   = j * 16 + n;
      q16[(size_t)row * HN + h] = (_Float16)acc[0 + j][r];
      k16[(size_t)row * HN + h] = (_Float16)acc[4 + j][r];
      const int b = row >> 12;        // T = 4096
      const int t = row & (TN - 1);
      vt16[((size_t)b * HN + h) * TN + t] = (_Float16)acc[8 + j][r];
    }
  }
}

// ---------------------------------------------------------------------------
// Kernel 3: flash attention with async-staged K/V tiles in LDS.
// Block = 8 waves, all on the same batch; waves share the K/V key-tile stream.
// Per 32-key tile: async copy K (32x64 f16, 4KB) and V^T (64x32 f16, 4KB)
// into double-buffered LDS (ASYNCcnt), one barrier per iteration; each wave
// computes S = Q*K^T (WMMA), online softmax, P restage via per-wave LDS,
// O += P*V (WMMA).
//
// Dynamic LDS layout (no static LDS -> dynamic base offset 0, required for
// raw LDS byte addresses in global_load_async_to_lds):
//   [    0,  8192)  K tiles,  2 buffers x 4096B, layout [t][h]
//   [ 8192, 16384)  V tiles,  2 buffers x 4096B, layout [h][t]
//   [16384, 24576)  per-wave P staging, 8 x 1024B
__global__ void __launch_bounds__(256)
flash_attn(const _Float16* __restrict__ q16, const _Float16* __restrict__ k16,
           const _Float16* __restrict__ vt16, const int* __restrict__ amask,
           float* __restrict__ out) {
  extern __shared__ char smem[];

  const int tid  = threadIdx.x;
  const int lane = tid & 31;
  const int wave = tid >> 5;
  const int g    = blockIdx.x * 8 + wave;    // global q-tile id, 0..1023
  const int b    = g >> 8;                   // 256 q-tiles per batch
  const int qt   = g & 255;

  const char* kgb = (const char*)(k16 + (size_t)b * TN * HN);    // [t][h]
  const char* vgb = (const char*)(vt16 + (size_t)b * HN * TN);   // [h][t]

  // cooperative async fetch of one 32-key K tile + V tile into LDS buffer
  auto fetch_kv = [&](int kt, int buf) {
    // K tile: contiguous 4KB; 256 threads x 16B
    uint32_t klds = (uint32_t)(buf * 4096 + tid * 16);
    uint32_t kgof = (uint32_t)(kt * (HN * 2) + tid * 16);
    asm volatile("global_load_async_to_lds_b128 %0, %1, %2"
                 :: "v"(klds), "v"(kgof), "s"(kgb) : "memory");
    // V tile: 64 rows of 64B, global row stride TN*2
    int h = tid >> 2, seg = tid & 3;
    uint32_t vlds = (uint32_t)(8192 + buf * 4096 + h * 64 + seg * 16);
    uint32_t vgof = (uint32_t)(h * (TN * 2) + kt * 2 + seg * 16);
    asm volatile("global_load_async_to_lds_b128 %0, %1, %2"
                 :: "v"(vlds), "v"(vgof), "s"(vgb) : "memory");
  };

  const _Float16* qb = q16 + ((size_t)b * TN + qt * 16) * HN;
  const v16h aq0 = load_tile_h(qb, HN, lane);        // h = 0..31
  const v16h aq1 = load_tile_h(qb + 32, HN, lane);   // h = 32..63

  v8f o[4];
#pragma unroll
  for (int j = 0; j < 4; ++j) o[j] = (v8f)0.0f;
  float m[8], l[8];
#pragma unroll
  for (int r = 0; r < 8; ++r) { m[r] = NEG_BIG; l[r] = 0.0f; }

  const int* am   = amask + (size_t)b * TN;
  _Float16* myp   = (_Float16*)(smem + 16384) + wave * (16 * 32);
  const int half  = lane >> 4;
  const int col   = lane & 15;

  fetch_kv(0, 0);
  asm volatile("s_wait_asynccnt 0x0" ::: "memory");
  __syncthreads();

  for (int kt = 0; kt < TN; kt += 32) {
    const int cur = (kt >> 5) & 1;
    if (kt + 32 < TN) fetch_kv(kt + 32, cur ^ 1);   // prefetch next tile

    const _Float16* kbufp = (const _Float16*)(smem + cur * 4096);
    const _Float16* vbufp = (const _Float16*)(smem + 8192 + cur * 4096);

    // key padding mask: one key column per lane per sub-tile
    const bool ok0 = am[kt + col] != 0;
    const bool ok1 = am[kt + 16 + col] != 0;

    // S tiles from LDS-staged K (B operand = K^T, column-major via [t][h])
    v8f s0 = (v8f)0.0f, s1 = (v8f)0.0f;
    s0 = wmma16(aq0, load_tile_h(kbufp, HN, lane), s0);
    s0 = wmma16(aq1, load_tile_h(kbufp + 32, HN, lane), s0);
    s1 = wmma16(aq0, load_tile_h(kbufp + 16 * HN, HN, lane), s1);
    s1 = wmma16(aq1, load_tile_h(kbufp + 16 * HN + 32, HN, lane), s1);

    // scale + mask, then per-row online-softmax update
    float p0[8], p1[8];
#pragma unroll
    for (int r = 0; r < 8; ++r) {
      const float a0 = ok0 ? s0[r] * SM_SCALE : NEG_BIG;
      const float a1 = ok1 ? s1[r] * SM_SCALE : NEG_BIG;
      float rmax = fmaxf(a0, a1);
#pragma unroll
      for (int off = 1; off <= 8; off <<= 1)
        rmax = fmaxf(rmax, __shfl_xor(rmax, off, 32));
      const float nm    = fmaxf(m[r], rmax);
      const float alpha = __expf(m[r] - nm);
      m[r] = nm;
      p0[r] = ok0 ? __expf(a0 - nm) : 0.0f;
      p1[r] = ok1 ? __expf(a1 - nm) : 0.0f;
      float rsum = p0[r] + p1[r];
#pragma unroll
      for (int off = 1; off <= 8; off <<= 1)
        rsum += __shfl_xor(rsum, off, 32);
      l[r] = l[r] * alpha + rsum;
#pragma unroll
      for (int j = 0; j < 4; ++j) o[j][r] *= alpha;
    }

    // P: C/D layout -> row-major LDS -> reload in A layout (same-wave, DScnt)
#pragma unroll
    for (int r = 0; r < 8; ++r) {
      const int row = r + 8 * half;
      myp[row * 32 + col]      = (_Float16)p0[r];
      myp[row * 32 + 16 + col] = (_Float16)p1[r];
    }
    asm volatile("s_wait_dscnt 0" ::: "memory");
    const v16h pa = load_tile_h(myp, 32, lane);

    // O += P * V from LDS-staged V^T ([h][t] => column-major B tile)
#pragma unroll
    for (int j = 0; j < 4; ++j) {
      v16h bv = load_tile_h(vbufp + (j * 16) * 32, 32, lane);
      o[j] = wmma16(pa, bv, o[j]);
    }

    // my prefetch landed + everyone done reading 'cur' before it is reused
    asm volatile("s_wait_asynccnt 0x0" ::: "memory");
    __syncthreads();
  }

  // epilogue: normalize by l and store f32 output [B,T,H]
  float* ob = out + ((size_t)b * TN + qt * 16) * HN;
#pragma unroll
  for (int r = 0; r < 8; ++r) {
    const float inv = 1.0f / l[r];
    const int row   = r + 8 * half;
#pragma unroll
    for (int j = 0; j < 4; ++j)
      ob[(size_t)row * HN + j * 16 + col] = o[j][r] * inv;
  }
}

// ---------------------------------------------------------------------------
extern "C" void kernel_launch(void* const* d_in, const int* in_sizes, int n_in,
                              void* d_out, int out_size, void* d_ws, size_t ws_size,
                              hipStream_t stream) {
  (void)in_sizes; (void)n_in; (void)out_size; (void)ws_size;
  const float* x  = (const float*)d_in[0];
  const int*   am = (const int*)d_in[1];
  const float* Wk = (const float*)d_in[2];
  const float* Wq = (const float*)d_in[3];
  const float* Wv = (const float*)d_in[4];
  float*       out = (float*)d_out;

  // workspace layout (f16): [3][H][C] weights^T, q16, k16, vt16
  _Float16* wt   = (_Float16*)d_ws;
  _Float16* q16  = wt  + (size_t)3 * HN * CN;
  _Float16* k16  = q16 + (size_t)BN * TN * HN;
  _Float16* vt16 = k16 + (size_t)BN * TN * HN;

  cvt_weights<<<(3 * CN * HN + 255) / 256, 256, 0, stream>>>(Wq, Wk, Wv, wt);
  qkv_proj<<<(BN * TN) / 128, 256, 0, stream>>>(x, wt, q16, k16, vt16);
  flash_attn<<<(BN * TN / 16) / 8, 256, 24576, stream>>>(q16, k16, vt16, am, out);
}
